// NNMFMixerEncoder_7181185318969
// MI455X (gfx1250) — matrix operand order repaired and checked by reading, not compiled, not run on hardware
//
#include <hip/hip_runtime.h>
#include <hip/hip_bf16.h>

// ---------------------------------------------------------------------------
// NNMF-mixer transformer encoder block on MI455X (gfx1250).
// bf16 WMMA (f32 accumulate) + async global->LDS tile staging (ASYNCcnt path).
// ---------------------------------------------------------------------------

typedef __attribute__((ext_vector_type(16))) __bf16 v16bf;
typedef __attribute__((ext_vector_type(8)))  __bf16 v8bf;
typedef __attribute__((ext_vector_type(8)))  float  v8f;

#define MIN_POS 1e-6f

static __device__ __forceinline__ v8f wmma_bf16(v16bf a, v16bf b, v8f c) {
  return __builtin_amdgcn_wmma_f32_16x16x32_bf16(false, a, false, b, (short)0, c,
                                                 false, false);
}

static __device__ __forceinline__ v8f vzero8f() {
  v8f z = {0.f, 0.f, 0.f, 0.f, 0.f, 0.f, 0.f, 0.f};
  return z;
}

// CDNA5 async DMA: global -> LDS, 16B per lane, tracked by ASYNCcnt.
// lds = LDS byte offset (low 32 bits of the shared-aperture flat address).
static __device__ __forceinline__ void async_b128(unsigned lds, const void* g) {
  asm volatile("global_load_async_to_lds_b128 %0, %1, off"
               :: "v"(lds), "v"(g) : "memory");
}
static __device__ __forceinline__ void async_wait0() {
  asm volatile("s_wait_asynccnt 0x0" ::: "memory");
}
static __device__ __forceinline__ unsigned lds_off(const void* p) {
  return (unsigned)(size_t)p;   // shared aperture: addr[31:0] == LDS offset
}

// A-matrix fragment (16x32 bf16, row-major source with leading dim ld):
// lane<16: M=lane, K in {0..7, 16..23}; lane>=16: M=lane-16, K in {8..15, 24..31}
static __device__ __forceinline__ v16bf load_a_frag(const __bf16* base, int ld, int lane) {
  int m  = lane & 15;
  int k0 = (lane < 16) ? 0 : 8;
  union { v16bf v; v8bf h[2]; } u;
  u.h[0] = *(const v8bf*)(base + m * ld + k0);
  u.h[1] = *(const v8bf*)(base + m * ld + k0 + 16);
  return u.v;
}

// B-matrix fragment (32x16 bf16, row-major source): lane<16: N=lane, K in {0..7, 16..23}; lane>=16: N=lane-16, K in {8..15, 24..31}
static __device__ __forceinline__ v16bf load_b_frag(const __bf16* base, int ld, int lane) {
  v16bf v;
  int n  = lane & 15;
  int k0 = (lane < 16) ? 0 : 8;
#pragma unroll
  for (int s = 0; s < 8; ++s) {
    v[s]     = base[(k0 + s) * ld + n];
    v[s + 8] = base[(k0 + 16 + s) * ld + n];
  }
  return v;
}

// ---------------------------------------------------------------------------
// Elementwise / prep kernels
// ---------------------------------------------------------------------------

__global__ void k_cvt(const float* __restrict__ s, __bf16* __restrict__ d, int n) {
  for (int i = blockIdx.x * blockDim.x + threadIdx.x; i < n; i += gridDim.x * blockDim.x)
    d[i] = (__bf16)s[i];
}

// dst[C][R] = (bf16) src[R][C]; grid = (C/32, R/32), 256 threads
__global__ __launch_bounds__(256)
void k_transpose_cvt(const float* __restrict__ s, __bf16* __restrict__ d, int R, int C) {
  __shared__ float tile[32][33];
  int bx = blockIdx.x * 32;   // src col
  int by = blockIdx.y * 32;   // src row
  int tx = threadIdx.x & 31, ty = threadIdx.x >> 5;
#pragma unroll
  for (int i = 0; i < 32; i += 8)
    tile[ty + i][tx] = s[(size_t)(by + ty + i) * C + bx + tx];
  __syncthreads();
#pragma unroll
  for (int i = 0; i < 32; i += 8)
    d[(size_t)(bx + ty + i) * R + by + tx] = (__bf16)tile[tx][ty + i];
}

// LayerNorm over N=512 columns; one block (256 thr) per row; bf16 output
__global__ __launch_bounds__(256)
void k_layernorm(const float* __restrict__ x, const float* __restrict__ g,
                 const float* __restrict__ b, __bf16* __restrict__ out) {
  const int N = 512;
  size_t base = (size_t)blockIdx.x * N;
  int tid = threadIdx.x;
  float v0 = x[base + tid], v1 = x[base + tid + 256];
  float s = v0 + v1, s2 = v0 * v0 + v1 * v1;
#pragma unroll
  for (int m = 16; m; m >>= 1) { s += __shfl_xor(s, m, 32); s2 += __shfl_xor(s2, m, 32); }
  __shared__ float ss[8], ss2[8];
  int wid = tid >> 5, lane = tid & 31;
  if (lane == 0) { ss[wid] = s; ss2[wid] = s2; }
  __syncthreads();
  if (wid == 0) {
    float a  = (lane < 8) ? ss[lane]  : 0.f;
    float a2 = (lane < 8) ? ss2[lane] : 0.f;
#pragma unroll
    for (int m = 4; m; m >>= 1) { a += __shfl_xor(a, m, 32); a2 += __shfl_xor(a2, m, 32); }
    if (lane == 0) { ss[0] = a; ss2[0] = a2; }
  }
  __syncthreads();
  float mu  = ss[0] * (1.0f / N);
  float var = ss2[0] * (1.0f / N) - mu * mu;
  float rs  = rsqrtf(var + 1e-5f);
  out[base + tid]       = (__bf16)((v0 - mu) * rs * g[tid] + b[tid]);
  out[base + tid + 256] = (__bf16)((v1 - mu) * rs * g[tid + 256] + b[tid + 256]);
}

// clip(.,MIN_POS) + per-head (64-wide) L1 normalize; one block per token row
__global__ __launch_bounds__(256)
void k_headnorm(const float* __restrict__ xe, __bf16* __restrict__ xn) {
  size_t base = (size_t)blockIdx.x * 512;
  int head = threadIdx.x >> 5, lane = threadIdx.x & 31;
  size_t o = base + head * 64 + lane * 2;
  float v0 = fmaxf(xe[o], MIN_POS), v1 = fmaxf(xe[o + 1], MIN_POS);
  float s = v0 + v1;
#pragma unroll
  for (int m = 16; m; m >>= 1) s += __shfl_xor(s, m, 32);
  float rn = 1.0f / fmaxf(s, 1e-12f);
  xn[o]     = (__bf16)(v0 * rn);
  xn[o + 1] = (__bf16)(v1 * rn);
}

// ---------------------------------------------------------------------------
// Generic bf16 WMMA GEMM: Out[M,N] = epi(A[M,K] * Bm[K,N] + bias)
// block tile 128x128, 8 waves (4 M x 2 N), wave tile 32x64, K step 32.
// Double-buffered LDS staging via async global->LDS DMA (ASYNCcnt).
// ---------------------------------------------------------------------------

enum { EPI_STORE_F32 = 0, EPI_RESID_F32 = 1, EPI_GELU_BF16 = 2 };

template <int EPI>
__global__ __launch_bounds__(256)
void k_gemm(const __bf16* __restrict__ A, int lda,
            const __bf16* __restrict__ Bm, int ldb,
            const float* __restrict__ bias, const float* __restrict__ resid,
            void* __restrict__ out, int ldo, int K) {
  __shared__ __attribute__((aligned(16))) __bf16 shA[2][128 * 40];
  __shared__ __attribute__((aligned(16))) __bf16 shB[2][32 * 136];
  int tid = threadIdx.x;
  int wid = tid >> 5, lane = tid & 31;
  int wm = wid & 3, wn = wid >> 2;
  int bm0 = blockIdx.y * 128, bn0 = blockIdx.x * 128;

  v8f acc[2][4];
#pragma unroll
  for (int i = 0; i < 2; ++i)
#pragma unroll
    for (int j = 0; j < 4; ++j) acc[i][j] = vzero8f();

  // per-thread staging coordinates
  int ra = tid >> 1, ca = (tid & 1) * 16;    // A: [128 x 32]
  int rb = tid >> 3, cb = (tid & 7) * 16;    // B: [32 x 128]

  auto stage = [&](int buf, int kk) {
    const __bf16* srcA = A + (size_t)(bm0 + ra) * lda + kk + ca;
    unsigned la = lds_off(&shA[buf][ra * 40 + ca]);
    async_b128(la, srcA);
    async_b128(la + 16, srcA + 8);
    const __bf16* srcB = Bm + (size_t)(kk + rb) * ldb + bn0 + cb;
    unsigned lb = lds_off(&shB[buf][rb * 136 + cb]);
    async_b128(lb, srcB);
    async_b128(lb + 16, srcB + 8);
  };

  stage(0, 0);
  async_wait0();
  __syncthreads();

#pragma unroll 4
  for (int kk = 0; kk < K; kk += 32) {
    int cur = (kk >> 5) & 1;        // ping-pong by K-step parity
    int nxt = cur ^ 1;
    if (kk + 32 < K) stage(nxt, kk + 32);     // DMA next chunk, overlaps WMMA
    v16bf a0 = load_a_frag(&shA[cur][(wm * 32) * 40], 40, lane);
    v16bf a1 = load_a_frag(&shA[cur][(wm * 32 + 16) * 40], 40, lane);
#pragma unroll
    for (int nt = 0; nt < 4; ++nt) {
      v16bf bb = load_b_frag(&shB[cur][wn * 64 + nt * 16], 136, lane);
      acc[0][nt] = wmma_bf16(a0, bb, acc[0][nt]);
      acc[1][nt] = wmma_bf16(a1, bb, acc[1][nt]);
    }
    if (kk + 32 < K) {
      async_wait0();
      __syncthreads();
    }
  }

  int half = lane >> 4, ln = lane & 15;
#pragma unroll
  for (int mt = 0; mt < 2; ++mt)
#pragma unroll
    for (int nt = 0; nt < 4; ++nt) {
      int col = bn0 + wn * 64 + nt * 16 + ln;
      float bv = bias[col];
#pragma unroll
      for (int r = 0; r < 8; ++r) {
        int row = bm0 + wm * 32 + mt * 16 + half * 8 + r;
        float v = acc[mt][nt][r] + bv;
        size_t oidx = (size_t)row * ldo + col;
        if (EPI == EPI_STORE_F32) {
          ((float*)out)[oidx] = v;
        } else if (EPI == EPI_RESID_F32) {
          ((float*)out)[oidx] = v + resid[oidx];
        } else {  // tanh-approx GELU, bf16 out
          float gl = 0.5f * v * (1.0f + tanhf(0.7978845608028654f *
                                              (v + 0.044715f * v * v * v)));
          ((__bf16*)out)[oidx] = (__bf16)gl;
        }
      }
    }
}

// ---------------------------------------------------------------------------
// NNMF loop kernels: one block per (batch, head); 8 waves, each 32 rows x 64 cols
// ---------------------------------------------------------------------------

// r1 = gw^T @ h ; r = L1norm(clip(r1 @ lw)) ; t = x / r
__global__ __launch_bounds__(256)
void k_nnmf_A(const __bf16* __restrict__ gwT, const __bf16* __restrict__ lw,
              const __bf16* __restrict__ h, const __bf16* __restrict__ xn,
              __bf16* __restrict__ rout, __bf16* __restrict__ tout) {
  __shared__ __attribute__((aligned(16))) __bf16 sh_h[256 * 72];
  __shared__ __attribute__((aligned(16))) __bf16 r1[256 * 72];
  int b = blockIdx.x >> 3, head = blockIdx.x & 7;
  size_t base = (size_t)b * 256 * 512 + head * 64;   // row stride = 512 bf16
  const __bf16* hB  = h + base;
  const __bf16* xnB = xn + base;
  __bf16* rB = rout + base;
  __bf16* tB = tout + base;
  int tid = threadIdx.x, wid = tid >> 5, lane = tid & 31;
  int i0 = wid * 32;
  int half = lane >> 4, ln = lane & 15;

  // ---- async-stage h tile [256 x 64] into LDS (read by all 8 waves) ----
#pragma unroll
  for (int p = 0; p < 8; ++p) {
    int idx = p * 256 + tid;          // 0..2047 chunks of 8 bf16
    int r = idx >> 3, c = (idx & 7) * 8;
    async_b128(lds_off(&sh_h[r * 72 + c]), hB + (size_t)r * 512 + c);
  }
  async_wait0();
  __syncthreads();

  // ---- phase 1: r1[i,g] = sum_o gwT[i,o] * h[o,g]  (K=256)
  v8f acc[2][4];
#pragma unroll
  for (int i = 0; i < 2; ++i)
#pragma unroll
    for (int j = 0; j < 4; ++j) acc[i][j] = vzero8f();
#pragma unroll
  for (int kk = 0; kk < 256; kk += 32) {
    v16bf a0 = load_a_frag(gwT + (size_t)i0 * 256 + kk, 256, lane);
    v16bf a1 = load_a_frag(gwT + (size_t)(i0 + 16) * 256 + kk, 256, lane);
#pragma unroll
    for (int nt = 0; nt < 4; ++nt) {
      v16bf bb = load_b_frag(&sh_h[kk * 72 + nt * 16], 72, lane);
      acc[0][nt] = wmma_bf16(a0, bb, acc[0][nt]);
      acc[1][nt] = wmma_bf16(a1, bb, acc[1][nt]);
    }
  }
#pragma unroll
  for (int mt = 0; mt < 2; ++mt)
#pragma unroll
    for (int nt = 0; nt < 4; ++nt)
#pragma unroll
      for (int r = 0; r < 8; ++r) {
        int row = i0 + mt * 16 + half * 8 + r;
        r1[row * 72 + nt * 16 + ln] = (__bf16)acc[mt][nt][r];
      }
  __syncthreads();

  // ---- phase 2: r2 = r1 @ lw  (K=64), then clip + row L1-norm, t = x / r
  v8f acc2[2][4];
#pragma unroll
  for (int i = 0; i < 2; ++i)
#pragma unroll
    for (int j = 0; j < 4; ++j) acc2[i][j] = vzero8f();
#pragma unroll
  for (int kk = 0; kk < 64; kk += 32) {
    v16bf a0 = load_a_frag(&r1[i0 * 72 + kk], 72, lane);
    v16bf a1 = load_a_frag(&r1[(i0 + 16) * 72 + kk], 72, lane);
#pragma unroll
    for (int nt = 0; nt < 4; ++nt) {
      v16bf bb = load_b_frag(lw + kk * 64 + nt * 16, 64, lane);
      acc2[0][nt] = wmma_bf16(a0, bb, acc2[0][nt]);
      acc2[1][nt] = wmma_bf16(a1, bb, acc2[1][nt]);
    }
  }
#pragma unroll
  for (int mt = 0; mt < 2; ++mt)
#pragma unroll
    for (int r = 0; r < 8; ++r) {
      float v[4], s = 0.f;
#pragma unroll
      for (int nt = 0; nt < 4; ++nt) { v[nt] = fmaxf(acc2[mt][nt][r], MIN_POS); s += v[nt]; }
#pragma unroll
      for (int m = 8; m; m >>= 1) s += __shfl_xor(s, m, 32);   // within 16-lane half
      float rn = 1.0f / fmaxf(s, 1e-12f);
      int row = i0 + mt * 16 + half * 8 + r;
#pragma unroll
      for (int nt = 0; nt < 4; ++nt) {
        int col = nt * 16 + ln;
        float rv = v[nt] * rn;
        size_t o = (size_t)row * 512 + col;
        rB[o] = (__bf16)rv;
        tB[o] = (__bf16)((float)xnB[o] / rv);
      }
    }
}

// u1 = t @ lw^T ; h = L1norm(clip(h * (gw @ u1)))
__global__ __launch_bounds__(256)
void k_nnmf_B(const __bf16* __restrict__ gw, const __bf16* __restrict__ lwT,
              const __bf16* __restrict__ t, __bf16* __restrict__ h) {
  __shared__ __attribute__((aligned(16))) __bf16 u1[256 * 72];
  int b = blockIdx.x >> 3, head = blockIdx.x & 7;
  size_t base = (size_t)b * 256 * 512 + head * 64;
  const __bf16* tB = t + base;
  __bf16* hB = h + base;
  int tid = threadIdx.x, wid = tid >> 5, lane = tid & 31;
  int i0 = wid * 32;
  int half = lane >> 4, ln = lane & 15;

  // ---- phase 1: u1[i,g] = sum_f t[i,f] * lwT[f,g]  (K=64)
  v8f acc[2][4];
#pragma unroll
  for (int i = 0; i < 2; ++i)
#pragma unroll
    for (int j = 0; j < 4; ++j) acc[i][j] = vzero8f();
#pragma unroll
  for (int kk = 0; kk < 64; kk += 32) {
    v16bf a0 = load_a_frag(tB + (size_t)i0 * 512 + kk, 512, lane);
    v16bf a1 = load_a_frag(tB + (size_t)(i0 + 16) * 512 + kk, 512, lane);
#pragma unroll
    for (int nt = 0; nt < 4; ++nt) {
      v16bf bb = load_b_frag(lwT + kk * 64 + nt * 16, 64, lane);
      acc[0][nt] = wmma_bf16(a0, bb, acc[0][nt]);
      acc[1][nt] = wmma_bf16(a1, bb, acc[1][nt]);
    }
  }
#pragma unroll
  for (int mt = 0; mt < 2; ++mt)
#pragma unroll
    for (int nt = 0; nt < 4; ++nt)
#pragma unroll
      for (int r = 0; r < 8; ++r) {
        int row = i0 + mt * 16 + half * 8 + r;
        u1[row * 72 + nt * 16 + ln] = (__bf16)acc[mt][nt][r];
      }
  __syncthreads();

  // ---- phase 2: u2[o,g] = sum_i gw[o,i] * u1[i,g]  (K=256), h update
  v8f acc2[2][4];
#pragma unroll
  for (int i = 0; i < 2; ++i)
#pragma unroll
    for (int j = 0; j < 4; ++j) acc2[i][j] = vzero8f();
#pragma unroll
  for (int kk = 0; kk < 256; kk += 32) {
    v16bf a0 = load_a_frag(gw + (size_t)i0 * 256 + kk, 256, lane);
    v16bf a1 = load_a_frag(gw + (size_t)(i0 + 16) * 256 + kk, 256, lane);
#pragma unroll
    for (int nt = 0; nt < 4; ++nt) {
      v16bf bb = load_b_frag(&u1[kk * 72 + nt * 16], 72, lane);
      acc2[0][nt] = wmma_bf16(a0, bb, acc2[0][nt]);
      acc2[1][nt] = wmma_bf16(a1, bb, acc2[1][nt]);
    }
  }
#pragma unroll
  for (int mt = 0; mt < 2; ++mt)
#pragma unroll
    for (int r = 0; r < 8; ++r) {
      float v[4], s = 0.f;
      int row = i0 + mt * 16 + half * 8 + r;
#pragma unroll
      for (int nt = 0; nt < 4; ++nt) {
        size_t o = (size_t)row * 512 + nt * 16 + ln;
        v[nt] = fmaxf((float)hB[o] * acc2[mt][nt][r], MIN_POS);
        s += v[nt];
      }
#pragma unroll
      for (int m = 8; m; m >>= 1) s += __shfl_xor(s, m, 32);
      float rn = 1.0f / fmaxf(s, 1e-12f);
#pragma unroll
      for (int nt = 0; nt < 4; ++nt)
        hB[(size_t)row * 512 + nt * 16 + ln] = (__bf16)(v[nt] * rn);
    }
}

// ---------------------------------------------------------------------------
// Host launcher
// ---------------------------------------------------------------------------

extern "C" void kernel_launch(void* const* d_in, const int* in_sizes, int n_in,
                              void* d_out, int out_size, void* d_ws, size_t ws_size,
                              hipStream_t stream) {
  (void)in_sizes; (void)n_in; (void)out_size; (void)ws_size;
  const int Bb = 128, S = 256, Fdim = 512, E = 512;
  const int M = Bb * S;                 // 32768 token rows
  const size_t MB = 1ull << 20;

  const float* x       = (const float*)d_in[0];
  const float* embed_w = (const float*)d_in[1];
  const float* embed_b = (const float*)d_in[2];
  const float* local_w = (const float*)d_in[3];
  const float* global_w= (const float*)d_in[4];
  const float* out_w   = (const float*)d_in[5];
  const float* out_b   = (const float*)d_in[6];
  const float* ln1_g   = (const float*)d_in[7];
  const float* ln1_b   = (const float*)d_in[8];
  const float* ln2_g   = (const float*)d_in[9];
  const float* ln2_b   = (const float*)d_in[10];
  const float* mlp_w1  = (const float*)d_in[11];
  const float* mlp_b1  = (const float*)d_in[12];
  const float* mlp_w2  = (const float*)d_in[13];
  const float* mlp_b2  = (const float*)d_in[14];
  const float* h0      = (const float*)d_in[15];

  char* ws = (char*)d_ws;
  __bf16* xln  = (__bf16*)(ws + 0 * MB);     // 32MB (reused as xln2)
  __bf16* xn   = (__bf16*)(ws + 32 * MB);    // 32MB
  __bf16* hbuf = (__bf16*)(ws + 64 * MB);    // 32MB
  __bf16* rbuf = (__bf16*)(ws + 96 * MB);    // 32MB
  float*  x2   = (float*) (ws + 128 * MB);   // 64MB
  // 192..320MB scratch region, time-shared: xe (f32,64MB) -> t (bf16,32MB) -> m1 (bf16,128MB)
  float*  xe   = (float*) (ws + 192 * MB);
  __bf16* tbuf = (__bf16*)(ws + 192 * MB);
  __bf16* m1   = (__bf16*)(ws + 192 * MB);
  char* wp = ws + 320 * MB;
  __bf16* embed_wT = (__bf16*)wp; wp += (size_t)Fdim * E * 2;
  __bf16* out_wT   = (__bf16*)wp; wp += (size_t)E * Fdim * 2;
  __bf16* w1T      = (__bf16*)wp; wp += (size_t)Fdim * 2048 * 2;
  __bf16* w2T      = (__bf16*)wp; wp += (size_t)2048 * Fdim * 2;
  __bf16* gw_bf    = (__bf16*)wp; wp += (size_t)S * S * 2;
  __bf16* gwT_bf   = (__bf16*)wp; wp += (size_t)S * S * 2;
  __bf16* lw_bf    = (__bf16*)wp; wp += (size_t)64 * 64 * 2;
  __bf16* lwT_bf   = (__bf16*)wp; wp += (size_t)64 * 64 * 2;

  // -------- weight prep (transpose/convert to bf16) --------
  k_transpose_cvt<<<dim3(Fdim/32, E/32),   256, 0, stream>>>(embed_w, embed_wT, E, Fdim);
  k_transpose_cvt<<<dim3(E/32,   Fdim/32), 256, 0, stream>>>(out_w,   out_wT, Fdim, E);
  k_transpose_cvt<<<dim3(Fdim/32, 2048/32),256, 0, stream>>>(mlp_w1,  w1T, 2048, Fdim);
  k_transpose_cvt<<<dim3(2048/32, Fdim/32),256, 0, stream>>>(mlp_w2,  w2T, Fdim, 2048);
  k_transpose_cvt<<<dim3(S/32, S/32),      256, 0, stream>>>(global_w, gwT_bf, S, S);
  k_transpose_cvt<<<dim3(64/32, 64/32),    256, 0, stream>>>(local_w,  lwT_bf, 64, 64);
  k_cvt<<<256, 256, 0, stream>>>(global_w, gw_bf, S * S);
  k_cvt<<<16,  256, 0, stream>>>(local_w,  lw_bf, 64 * 64);
  k_cvt<<<4096, 256, 0, stream>>>(h0, hbuf, M * E);

  // -------- LN1 + embed + head-wise L1 norm --------
  k_layernorm<<<M, 256, 0, stream>>>(x, ln1_g, ln1_b, xln);
  k_gemm<EPI_STORE_F32><<<dim3(E/128, M/128), 256, 0, stream>>>(
      xln, Fdim, embed_wT, E, embed_b, nullptr, xe, E, Fdim);
  k_headnorm<<<M, 256, 0, stream>>>(xe, xn);

  // -------- 10 NNMF iterations --------
  for (int it = 0; it < 10; ++it) {
    k_nnmf_A<<<Bb * 8, 256, 0, stream>>>(gwT_bf, lw_bf, hbuf, xn, rbuf, tbuf);
    k_nnmf_B<<<Bb * 8, 256, 0, stream>>>(gw_bf, lwT_bf, tbuf, hbuf);
  }

  // -------- output projection + residual --------
  k_gemm<EPI_RESID_F32><<<dim3(Fdim/128, M/128), 256, 0, stream>>>(
      rbuf, E, out_wT, Fdim, out_b, x, x2, Fdim, E);

  // -------- LN2 + MLP (gelu) + residual into d_out --------
  k_layernorm<<<M, 256, 0, stream>>>(x2, ln2_g, ln2_b, xln);
  k_gemm<EPI_GELU_BF16><<<dim3(2048/128, M/128), 256, 0, stream>>>(
      xln, Fdim, w1T, 2048, mlp_b1, nullptr, m1, 2048, Fdim);
  k_gemm<EPI_RESID_F32><<<dim3(Fdim/128, M/128), 256, 0, stream>>>(
      m1, 2048, w2T, Fdim, mlp_b2, x2, (float*)d_out, Fdim, 2048);
}
